// ScaledDotProductAttention_26216480375072
// MI455X (gfx1250) — matrix operand order repaired and checked by reading, not compiled, not run on hardware
//
#include <hip/hip_runtime.h>
#include <hip/hip_bf16.h>
#include <math.h>

typedef __attribute__((ext_vector_type(16))) _Float16 v16h;
typedef __attribute__((ext_vector_type(8)))  float    v8f;

#define SDPA_B 2
#define SDPA_H 16
#define SDPA_S 2048
#define SDPA_D 64
#define INV_SCALE 0.125f   // 1 / sqrt(64) per reference SCALE=8

// LDS: 16 rows x 2048 cols of f32 scores (128 KB) + reductions.
#define SC_STRIDE 2048
#define LDS_FLOATS (16 * SC_STRIDE + 16 * 8 + 16 + 16)

// One workgroup = one (b*h, q-tile of 16 rows). 4 waves x 32 lanes.
__global__ void sdpa_causal_wmma_kernel(const float* __restrict__ q,
                                        const float* __restrict__ kmat,
                                        const float* __restrict__ vmat,
                                        float* __restrict__ attn,
                                        float* __restrict__ out) {
    extern __shared__ float smem[];
    float* sc     = smem;                    // [16][SC_STRIDE]
    float* red    = smem + 16 * SC_STRIDE;   // [16][8]
    float* rowmax = red + 16 * 8;            // [16]
    float* rowinv = rowmax + 16;             // [16]

    const int qtile = blockIdx.x;            // 0..127
    const int bh    = blockIdx.y;            // 0..31
    const int tid   = threadIdx.x;
    const int lane  = tid & 31;              // wave32
    const int wave  = tid >> 5;              // 0..3

    const int m  = lane & 15;                // matrix row / N-col within tile
    const int hf = lane >> 4;                // lane half (0/1)

    const int Lvalid = (qtile + 1) * 16;             // causal row length
    const int Lpad   = (Lvalid + 31) & ~31;          // pad to K=32 chunks

    const size_t baseQKV = (size_t)bh * SDPA_S * SDPA_D;
    const float* qbase = q    + baseQKV + (size_t)qtile * 16 * SDPA_D;
    const float* kbase = kmat + baseQKV;
    const float* vbase = vmat + baseQKV;

    // ---------------- Phase 1: scores = (Q K^T)/8, masked, into LDS ----------
    // A fragment (Q rows qtile*16..+15): lane m = row, element e -> dim
    //   d = 16*hf + e; a0: d+0, a1: d+32
    v16h a0, a1;
    {
        const float* qr = qbase + (size_t)m * SDPA_D;
        #pragma unroll
        for (int v = 0; v < 8; ++v) {
            const int d0 = 16 * hf + 2 * v;
            a0[2 * v]     = (_Float16)qr[d0];
            a0[2 * v + 1] = (_Float16)qr[d0 + 1];
            a1[2 * v]     = (_Float16)qr[d0 + 32];
            a1[2 * v + 1] = (_Float16)qr[d0 + 33];
        }
    }

    for (int j = wave; j <= qtile; j += 4) {
        // B fragment = K^T tile: lane n = key col, VGPR pair v holds
        // contraction rows dd = 16*hf + 2v (+1); b1 covers dims 32..63.
        const float* kr = kbase + (size_t)(j * 16 + m) * SDPA_D;
        v16h b0, b1;
        #pragma unroll
        for (int v = 0; v < 8; ++v) {
            const int dd = 16 * hf + 2 * v;
            b0[2 * v]     = (_Float16)kr[dd];
            b0[2 * v + 1] = (_Float16)kr[dd + 1];
            b1[2 * v]     = (_Float16)kr[dd + 32];
            b1[2 * v + 1] = (_Float16)kr[dd + 33];
        }
        v8f c = {};
        c = __builtin_amdgcn_wmma_f32_16x16x32_f16(false, a0, false, b0,
                                                   (short)0, c, false, false);
        c = __builtin_amdgcn_wmma_f32_16x16x32_f16(false, a1, false, b1,
                                                   (short)0, c, false, false);
        // C layout: VGPR r -> row r + 8*hf, lane%16 -> col
        #pragma unroll
        for (int r = 0; r < 8; ++r) {
            const int row  = r + 8 * hf;
            const int kcol = j * 16 + m;
            float s = c[r] * INV_SCALE;
            if (kcol > qtile * 16 + row) s = -__builtin_inff();  // causal mask
            sc[row * SC_STRIDE + kcol] = s;
        }
    }
    __syncthreads();

    // ---------------- Phase 2: softmax over valid cols ----------------------
    {
        const int row = tid >> 3;
        const int sub = tid & 7;
        float lm = -__builtin_inff();
        for (int cix = sub; cix < Lvalid; cix += 8)
            lm = fmaxf(lm, sc[row * SC_STRIDE + cix]);
        red[row * 8 + sub] = lm;
        __syncthreads();
        if (sub == 0) {
            float mx = red[row * 8];
            #pragma unroll
            for (int i = 1; i < 8; ++i) mx = fmaxf(mx, red[row * 8 + i]);
            rowmax[row] = mx;
        }
        __syncthreads();
        const float mx = rowmax[row];
        float ls = 0.0f;
        for (int cix = sub; cix < Lvalid; cix += 8) {
            const float e = __expf(sc[row * SC_STRIDE + cix] - mx);
            sc[row * SC_STRIDE + cix] = e;
            ls += e;
        }
        red[row * 8 + sub] = ls;
        __syncthreads();
        if (sub == 0) {
            float sm = 0.0f;
            #pragma unroll
            for (int i = 0; i < 8; ++i) sm += red[row * 8 + i];
            rowinv[row] = 1.0f / sm;
        }
        __syncthreads();
    }

    // Normalize in LDS, zero the masked tail, stream attn rows (NT stores:
    // 512 MB of probabilities, zero reuse, keep it out of L2).
    {
        float* arow = attn + ((size_t)bh * SDPA_S + (size_t)qtile * 16) * SDPA_S;
        for (int idx = tid; idx < 16 * SDPA_S; idx += 128) {
            const int r2 = idx >> 11;          // S == 2048
            const int cc = idx & (SDPA_S - 1);
            float p = 0.0f;
            if (cc < Lvalid) p = sc[r2 * SC_STRIDE + cc] * rowinv[r2];
            __builtin_nontemporal_store(p, arow + (size_t)r2 * SDPA_S + cc);
            if (cc < Lpad) sc[r2 * SC_STRIDE + cc] = p;  // zero-pad tail chunk
        }
    }
    __syncthreads();

    // ---------------- Phase 3: out = P @ V (wave w owns d-cols w*16..+15) ----
    {
        v8f acc = {};
        for (int c0 = 0; c0 < Lpad; c0 += 32) {
            v16h ap, bv;
            #pragma unroll
            for (int v = 0; v < 8; ++v) {
                const int ko = 16 * hf + 2 * v;
                ap[2 * v]     = (_Float16)sc[m * SC_STRIDE + c0 + ko];
                ap[2 * v + 1] = (_Float16)sc[m * SC_STRIDE + c0 + ko + 1];
                const int key = c0 + 16 * hf + 2 * v;
                const float* vr = vbase + (size_t)key * SDPA_D + wave * 16 + m;
                bv[2 * v]     = (_Float16)vr[0];
                bv[2 * v + 1] = (_Float16)vr[SDPA_D];
            }
            acc = __builtin_amdgcn_wmma_f32_16x16x32_f16(false, ap, false, bv,
                                                         (short)0, acc, false, false);
        }
        float* orow = out + baseQKV + (size_t)qtile * 16 * SDPA_D;
        #pragma unroll
        for (int r = 0; r < 8; ++r) {
            const int row = r + 8 * hf;
            orow[(size_t)row * SDPA_D + wave * 16 + m] = acc[r];
        }
    }
}

extern "C" void kernel_launch(void* const* d_in, const int* in_sizes, int n_in,
                              void* d_out, int out_size, void* d_ws, size_t ws_size,
                              hipStream_t stream) {
    (void)in_sizes; (void)n_in; (void)d_ws; (void)ws_size; (void)out_size;
    const float* q = (const float*)d_in[0];
    const float* k = (const float*)d_in[1];
    const float* v = (const float*)d_in[2];
    // d_in[3] = mask (bool causal) — implied by kernel structure, unused.

    float* attn = (float*)d_out;                                   // B*H*S*S
    float* out  = attn + (size_t)SDPA_B * SDPA_H * SDPA_S * SDPA_S; // B*H*S*D

    dim3 grid(SDPA_S / 16, SDPA_B * SDPA_H);
    dim3 block(128);
    const size_t smem = LDS_FLOATS * sizeof(float);   // ~131.7 KB (< 320 KB/WGP)
    sdpa_causal_wmma_kernel<<<grid, block, smem, stream>>>(q, k, v, attn, out);
}
